// RelativeInformationAttention_31817117729140
// MI455X (gfx1250) — matrix operand order repaired and checked by reading, not compiled, not run on hardware
//
#include <hip/hip_runtime.h>

typedef __attribute__((ext_vector_type(16))) __bf16 v16bf;
typedef __attribute__((ext_vector_type(8)))  float  v8f;

#define B_   2
#define H_   8
#define S_   2048
#define D_   64
#define R_   128
#define NR_  257            // 2R+1
#define QT_  128            // queries per workgroup (8 waves x 16 rows)
#define NT_  64             // keys per main-loop iteration (4 col tiles)
#define NWAVES_ 8
#define QRS_ 260            // padded qr row stride (bf16 elements)

// LDS layout (bytes)
#define QR_BYTES   (QT_*QRS_*2)             // 66560   bf16 bias table
#define K_BYTES    (2*NT_*D_*2)             // 16384   bf16 K tiles (double buffered)
#define V_BYTES    (2*D_*NT_*2)             // 16384   bf16 V^T tiles (double buffered)
#define P_BYTES    (NWAVES_*16*NT_*2)       // 16384   bf16 P bounce tiles (per wave)
#define SMEM_BYTES (QR_BYTES+K_BYTES+V_BYTES+P_BYTES)   // 115712 -> 2 WGs / WGP

static __device__ __forceinline__ v8f wmma_bf16(v16bf a, v16bf b, v8f c) {
  // D(f32 16x16) = A(bf16 16x32) * B(bf16 32x16) + C
  return __builtin_amdgcn_wmma_f32_16x16x32_bf16(false, a, false, b, (short)0, c,
                                                 false, false);
}

__global__ __launch_bounds__(256, 1)
void relattn_kernel(const float* __restrict__ q, const float* __restrict__ k,
                    const float* __restrict__ v, const float* __restrict__ rel,
                    const int* __restrict__ qinfo, const int* __restrict__ kinfo,
                    float* __restrict__ out) {
  extern __shared__ __align__(32) char smem[];
  __bf16* qr_lds = (__bf16*)smem;                                   // [QT_][QRS_]
  __bf16* k_lds  = (__bf16*)(smem + QR_BYTES);                      // [2][NT_][D_]
  __bf16* vt_lds = (__bf16*)(smem + QR_BYTES + K_BYTES);            // [2][D_][NT_]
  __bf16* p_lds  = (__bf16*)(smem + QR_BYTES + K_BYTES + V_BYTES);  // [NWAVES_][16][NT_]

  const int tid  = threadIdx.x;
  const int wave = tid >> 5;
  const int lane = tid & 31;
  const int lh   = lane & 15;   // column (N) / row (M) index inside a 16-tile
  const int hi   = lane >> 4;   // which half of the wave

  const int bh    = blockIdx.y;         // b*H + h
  const int b     = bh / H_;
  const int qbase = blockIdx.x * QT_;
  const int qrow0 = qbase + wave * 16;  // this wave's 16-query strip

  // ---- Q fragments, loaded straight from global in A-matrix (16x32 bf16) layout ----
  const float* qrow = q + ((size_t)bh * S_ + qrow0 + lh) * D_;
  v16bf qf[2];
  #pragma unroll
  for (int f = 0; f < 2; ++f) {
    #pragma unroll
    for (int j = 0; j < 8; ++j) {
      qf[f][j]     = (__bf16)qrow[f*32 + hi*16 + j];
      qf[f][8 + j] = (__bf16)qrow[f*32 + hi*16 + 8 + j];
    }
  }

  // ---- per-row adjusted q_info (loop invariant) ----
  int qa[8];
  #pragma unroll
  for (int i = 0; i < 8; ++i) {
    int qi = qinfo[(size_t)b * S_ + qrow0 + i + hi*8];
    qa[i] = (qi == -1) ? 2 * R_ : qi;
  }

  // ---- bias table: qr[m][r] = sum_d Q[m][d] * rel[r][d]  (wave-private rows) ----
  for (int rt = 0; rt < (NR_ + 15) / 16; ++rt) {
    const int col = rt * 16 + lh;
    const int rc  = col < NR_ ? col : NR_ - 1;
    const float* rrow = rel + (size_t)rc * D_;
    v8f acc = {0.f,0.f,0.f,0.f,0.f,0.f,0.f,0.f};
    #pragma unroll
    for (int f = 0; f < 2; ++f) {
      v16bf bb;                                  // B-matrix 32x16: col=lh, dims contiguous
      #pragma unroll
      for (int j = 0; j < 16; ++j) bb[j] = (__bf16)rrow[f*32 + hi*16 + j];
      acc = wmma_bf16(qf[f], bb, acc);
    }
    #pragma unroll
    for (int i = 0; i < 8; ++i)
      if (col < NR_) qr_lds[(wave*16 + i + hi*8) * QRS_ + col] = (__bf16)acc[i];
  }

  // ---- K/V staging: each thread owns 16 consecutive dims of one key ----
  const int keyL = tid >> 2;           // 0..63
  const int dcol = (tid & 3) * 16;     // 0,16,32,48  (32B-aligned in bf16)
  float kreg[16], vreg[16];
  {
    const size_t g = ((size_t)bh * S_ + keyL) * D_ + dcol;
    #pragma unroll
    for (int j = 0; j < 16; ++j) { kreg[j] = k[g + j]; vreg[j] = v[g + j]; }
    v16bf kp;
    #pragma unroll
    for (int j = 0; j < 16; ++j) kp[j] = (__bf16)kreg[j];
    *(v16bf*)(k_lds + keyL * D_ + dcol) = kp;               // vector LDS store
    #pragma unroll
    for (int j = 0; j < 16; ++j)
      vt_lds[(dcol + j) * NT_ + keyL] = (__bf16)vreg[j];    // transposed scatter
  }
  __syncthreads();

  // ---- online-softmax state (C-layout rows: row = i + hi*8 of the strip) ----
  // m: row max (kept uniform across the 16 lanes of the row via shuffles)
  // lp: PER-LANE partial softmax denominator; reduced across lanes only at the end.
  float m[8], lp[8];
  v8f o[4];
  #pragma unroll
  for (int dt = 0; dt < 4; ++dt) o[dt] = (v8f){0.f,0.f,0.f,0.f,0.f,0.f,0.f,0.f};
  #pragma unroll
  for (int i = 0; i < 8; ++i) { m[i] = -1e30f; lp[i] = 0.f; }

  const int NKT = S_ / NT_;            // 32 iterations
  for (int kt = 0; kt < NKT; ++kt) {
    const int buf = kt & 1;
    const __bf16* kb = k_lds  + buf * NT_ * D_;
    const __bf16* vb = vt_lds + buf * D_ * NT_;

    // prefetch next tile into registers (overlaps with compute)
    if (kt + 1 < NKT) {
      const size_t g = ((size_t)bh * S_ + (kt + 1) * NT_ + keyL) * D_ + dcol;
      #pragma unroll
      for (int j = 0; j < 16; ++j) { kreg[j] = k[g + j]; vreg[j] = v[g + j]; }
    }

    // ---- scores: S = Q * K^T  (four 16-key column tiles, K-dim split 2x32) ----
    v8f s[4];
    #pragma unroll
    for (int ct = 0; ct < 4; ++ct) s[ct] = (v8f){0.f,0.f,0.f,0.f,0.f,0.f,0.f,0.f};
    #pragma unroll
    for (int f = 0; f < 2; ++f) {
      #pragma unroll
      for (int ct = 0; ct < 4; ++ct) {
        v16bf bk = *(const v16bf*)(kb + (ct*16 + lh) * D_ + f*32 + hi*16);
        s[ct] = wmma_bf16(qf[f], bk, s[ct]);
      }
    }

    // ---- relative bias gather + scale + online softmax ----
    const int kbase = kt * NT_;
    int ka[4];
    #pragma unroll
    for (int ct = 0; ct < 4; ++ct) {
      int kv = kinfo[(size_t)b * S_ + kbase + ct*16 + lh];
      ka[ct] = (kv == -1) ? 3 * R_ : kv;
    }

    __bf16* pw = p_lds + wave * 16 * NT_;
    #pragma unroll
    for (int i = 0; i < 8; ++i) {
      const int row = wave*16 + i + hi*8;
      float sc[4];
      #pragma unroll
      for (int ct = 0; ct < 4; ++ct) {
        int d = ka[ct] - qa[i];
        d = d < -R_ ? -R_ : (d > R_ ? R_ : d);
        sc[ct] = (s[ct][i] + (float)qr_lds[row * QRS_ + d + R_]) * 0.125f; // 1/sqrt(64)
      }

      float tmax = fmaxf(fmaxf(sc[0], sc[1]), fmaxf(sc[2], sc[3]));
      #pragma unroll
      for (int off = 8; off >= 1; off >>= 1)
        tmax = fmaxf(tmax, __shfl_xor(tmax, off, 32));
      const float mnew  = fmaxf(m[i], tmax);       // uniform across the row's 16 lanes
      const float scale = __expf(m[i] - mnew);     // uniform -> per-lane partials OK
      float p[4], prs = 0.f;
      #pragma unroll
      for (int ct = 0; ct < 4; ++ct) { p[ct] = __expf(sc[ct] - mnew); prs += p[ct]; }
      lp[i] = lp[i] * scale + prs;                 // per-lane partial; no shuffle here
      m[i] = mnew;
      #pragma unroll
      for (int dt = 0; dt < 4; ++dt) o[dt][i] *= scale;

      #pragma unroll
      for (int ct = 0; ct < 4; ++ct)
        pw[(i + hi*8) * NT_ + ct*16 + lh] = (__bf16)p[ct];   // C-layout -> row-major
    }
    asm volatile("s_wait_dscnt 0x0" ::: "memory");  // P stores visible before A-frag read

    // ---- O += P * V  (two A-fragments of P, K=32 keys each; B = V^T tiles) ----
    #pragma unroll
    for (int kf = 0; kf < 2; ++kf) {
      v16bf pa;
      #pragma unroll
      for (int j = 0; j < 8; ++j) {
        pa[j]     = pw[lh * NT_ + kf*32 + hi*16 + j];
        pa[8 + j] = pw[lh * NT_ + kf*32 + hi*16 + 8 + j];
      }
      #pragma unroll
      for (int dt = 0; dt < 4; ++dt) {
        v16bf bv = *(const v16bf*)(vb + (dt*16 + lh) * NT_ + kf*32 + hi*16);
        o[dt] = wmma_bf16(pa, bv, o[dt]);
      }
    }

    // ---- rotate double buffer ----
    if (kt + 1 < NKT) {
      __syncthreads();
      const int nb = (kt + 1) & 1;
      v16bf kp;
      #pragma unroll
      for (int j = 0; j < 16; ++j) kp[j] = (__bf16)kreg[j];
      *(v16bf*)(k_lds + nb * NT_ * D_ + keyL * D_ + dcol) = kp;
      #pragma unroll
      for (int j = 0; j < 16; ++j)
        vt_lds[nb * D_ * NT_ + (dcol + j) * NT_ + keyL] = (__bf16)vreg[j];
      __syncthreads();
    }
  }

  // ---- epilogue: finish deferred l reduction, normalize, store fp32 ----
  #pragma unroll
  for (int i = 0; i < 8; ++i) {
    float l = lp[i];
    #pragma unroll
    for (int off = 8; off >= 1; off >>= 1) l += __shfl_xor(l, off, 32);
    const float inv = 1.0f / l;
    const size_t obase = ((size_t)bh * S_ + qbase + wave*16 + i + hi*8) * D_;
    #pragma unroll
    for (int dt = 0; dt < 4; ++dt)
      out[obase + dt*16 + lh] = o[dt][i] * inv;
  }
}

extern "C" void kernel_launch(void* const* d_in, const int* in_sizes, int n_in,
                              void* d_out, int out_size, void* d_ws, size_t ws_size,
                              hipStream_t stream) {
  const float* q   = (const float*)d_in[0];
  const float* k   = (const float*)d_in[1];
  const float* v   = (const float*)d_in[2];
  const float* rel = (const float*)d_in[3];
  const int*   qi  = (const int*)d_in[4];
  const int*   ki  = (const int*)d_in[5];
  float* out = (float*)d_out;

  dim3 grid(S_ / QT_, B_ * H_);   // (16, 16)
  relattn_kernel<<<grid, 256, SMEM_BYTES, stream>>>(q, k, v, rel, qi, ki, out);

  (void)in_sizes; (void)n_in; (void)out_size; (void)d_ws; (void)ws_size;
}
